// CarbonCreditGNN_8778913153102
// MI455X (gfx1250) — matrix operand order repaired and checked by reading, not compiled, not run on hardware
//
#include <hip/hip_runtime.h>
#include <hip/hip_bf16.h>
#include <stdint.h>

// ---------------------------------------------------------------------------
// CarbonCreditGNN forward for MI455X (gfx1250), wave32 + WMMA bf16 +
// async global->LDS tile staging.
// ---------------------------------------------------------------------------

typedef __attribute__((ext_vector_type(16))) __bf16 v16bf;
typedef __attribute__((ext_vector_type(8)))  float  v8f;

#define NN      50000
#define NE      800000
#define ETOT    (NE + NN)       // edges + self loops
#define HID     128
#define HEADS   4
#define NEG_SLOPE 0.2f

// ---------------------------------------------------------------------------
// bf16 WMMA GEMM: C[M,N] = A[M,K] * B[K,N], where B is supplied TRANSPOSED
// (Bt[N,K] row-major) so both LDS tiles and both fragments are contiguous
// 128-bit runs. 64x64 block tile, 4 waves, wave -> 16 rows x 64 cols.
// Requires K % 32 == 0, N % 64 == 0 (all call sites satisfy this).
// Tiles staged with GLOBAL_LOAD_ASYNC_TO_LDS_B128 (ASYNCcnt) when the block
// is interior in M; edge blocks use guarded zero-filling loads.
// ---------------------------------------------------------------------------
#define APITCH 40   // bf16 elems per LDS row (80B, 16B aligned)

__global__ __launch_bounds__(128) void gemm_bf16_wmma(
    const __bf16* __restrict__ A,   // [M,K] row-major
    const __bf16* __restrict__ Bt,  // [N,K] row-major (B transposed)
    float* __restrict__ C, int M, int N, int K,
    const float* __restrict__ bias, int relu)
{
    __shared__ __align__(16) __bf16 As[64 * APITCH];
    __shared__ __align__(16) __bf16 Bs[64 * APITCH];

    const int tid  = threadIdx.x;
    const int wave = tid >> 5;
    const int lane = tid & 31;
    const int m    = lane & 15;     // row-in-16 (A) / col-in-16 (B,C)
    const int hf   = lane >> 4;     // lane half

    const int bN = blockIdx.x * 64;
    const int bM = blockIdx.y * 64;
    const bool interior = (bM + 64 <= M);

    v8f acc[4] = {};

    for (int kt = 0; kt < K; kt += 32) {
        // ---- stage A tile (64 rows x 32 k) ----
        if (interior) {
            for (int t = tid; t < 256; t += 128) {
                int row = t >> 2, q = t & 3;
                unsigned la = (unsigned)(uintptr_t)(As + row * APITCH + q * 8);
                const __bf16* ga = A + (size_t)(bM + row) * K + kt + q * 8;
                asm volatile("global_load_async_to_lds_b128 %0, %1, off"
                             :: "v"(la), "v"(ga) : "memory");
            }
        } else {
            for (int t = tid; t < 256; t += 128) {
                int row = t >> 2, q = t & 3;
                int gr = bM + row;
                uint4 v = {0u, 0u, 0u, 0u};
                if (gr < M) v = *(const uint4*)(A + (size_t)gr * K + kt + q * 8);
                *(uint4*)(As + row * APITCH + q * 8) = v;
            }
        }
        // ---- stage B tile (64 n-rows x 32 k) ----
        for (int t = tid; t < 256; t += 128) {
            int row = t >> 2, q = t & 3;
            unsigned lb = (unsigned)(uintptr_t)(Bs + row * APITCH + q * 8);
            const __bf16* gb = Bt + (size_t)(bN + row) * K + kt + q * 8;
            asm volatile("global_load_async_to_lds_b128 %0, %1, off"
                         :: "v"(lb), "v"(gb) : "memory");
        }
        asm volatile("s_wait_asynccnt 0x0" ::: "memory");
        __syncthreads();

        // ---- A fragment: one contiguous 32B run per lane ----
        v16bf afr;
        const __bf16* ar = As + (wave * 16 + m) * APITCH;
#pragma unroll
        for (int e = 0; e < 16; ++e)
            afr[e] = ar[(hf << 4) + e];

        // ---- B fragments: one contiguous 32B run per lane per subtile ----
#pragma unroll
        for (int ct = 0; ct < 4; ++ct) {
            v16bf bfr;
            const __bf16* br = Bs + (ct * 16 + m) * APITCH + (hf << 4);
#pragma unroll
            for (int e = 0; e < 16; ++e) bfr[e] = br[e];
            acc[ct] = __builtin_amdgcn_wmma_f32_16x16x32_bf16(
                false, afr, false, bfr, (short)0, acc[ct], false, false);
        }
        __syncthreads();
    }

    // ---- epilogue ----
#pragma unroll
    for (int ct = 0; ct < 4; ++ct)
#pragma unroll
        for (int r = 0; r < 8; ++r) {
            int gr = bM + wave * 16 + hf * 8 + r;
            int gc = bN + ct * 16 + m;
            if (gr < M) {
                float v = acc[ct][r];
                if (bias) v += bias[gc];
                if (relu && v < 0.f) v = 0.f;
                C[(size_t)gr * N + gc] = v;
            }
        }
}

// ---------------------------------------------------------------------------
// Elementwise / attention kernels
// ---------------------------------------------------------------------------
__global__ void cvt_bf16(const float* __restrict__ in, __bf16* __restrict__ out, long n)
{
    long i = (long)blockIdx.x * blockDim.x + threadIdx.x;
    if (i < n) out[i] = (__bf16)in[i];
}

// out[n*K + k] = (bf16) in[k*N + n]   (weight transpose, tiny matrices)
__global__ void cvt_t_bf16(const float* __restrict__ in, __bf16* __restrict__ out,
                           int K, int N)
{
    long i = (long)blockIdx.x * blockDim.x + threadIdx.x;
    if (i >= (long)K * N) return;
    int n = (int)(i / K), k = (int)(i % K);
    out[i] = (__bf16)in[(size_t)k * N + n];
}

__global__ void fill_u32(uint32_t* __restrict__ p, uint32_t v, long n)
{
    long i = (long)blockIdx.x * blockDim.x + threadIdx.x;
    if (i < n) p[i] = v;
}

// monotonic order-preserving float<->uint mapping (for atomicMax)
__device__ __forceinline__ unsigned enc_f32(float f)
{
    unsigned u = __float_as_uint(f);
    return (u & 0x80000000u) ? ~u : (u | 0x80000000u);
}
__device__ __forceinline__ float dec_f32(unsigned u)
{
    u = (u & 0x80000000u) ? (u & 0x7fffffffu) : ~u;
    return __uint_as_float(u);
}
#define ENC_NEG_INF 0x007FFFFFu   // enc_f32(-inf)

__global__ void attn_logits(const float* __restrict__ h,
                            const float* __restrict__ a_s, const float* __restrict__ a_d,
                            float* __restrict__ als, float* __restrict__ ald,
                            int Nn, int H, int D)
{
    long i = (long)blockIdx.x * blockDim.x + threadIdx.x;   // node*H + head
    if (i >= (long)Nn * H) return;
    int hh = (int)(i % H);
    const float* hp = h + (i / H) * (size_t)(H * D) + (size_t)hh * D;
    const float* sp = a_s + (size_t)hh * D;
    const float* dp = a_d + (size_t)hh * D;
    float s = 0.f, d = 0.f;
    for (int k = 0; k < D; ++k) { float v = hp[k]; s += v * sp[k]; d += v * dp[k]; }
    als[i] = s; ald[i] = d;
}

__global__ void edge_max(const int* __restrict__ row, const int* __restrict__ col,
                         const float* __restrict__ als, const float* __restrict__ ald,
                         unsigned* __restrict__ mmax, int E, int ET, int H)
{
    long i = (long)blockIdx.x * blockDim.x + threadIdx.x;
    if (i >= (long)ET * H) return;
    int e = (int)(i / H), hh = (int)(i % H);
    int r, c;
    if (e < E) { r = row[e]; c = col[e]; } else { r = c = e - E; }
    float v = als[(size_t)r * H + hh] + ald[(size_t)c * H + hh];
    v = (v > 0.f) ? v : NEG_SLOPE * v;
    atomicMax(&mmax[(size_t)c * H + hh], enc_f32(v));
}

__global__ void edge_exp(const int* __restrict__ row, const int* __restrict__ col,
                         const float* __restrict__ als, const float* __restrict__ ald,
                         const unsigned* __restrict__ mmax,
                         float* __restrict__ ex, float* __restrict__ ssum,
                         int E, int ET, int H)
{
    long i = (long)blockIdx.x * blockDim.x + threadIdx.x;
    if (i >= (long)ET * H) return;
    int e = (int)(i / H), hh = (int)(i % H);
    int r, c;
    if (e < E) { r = row[e]; c = col[e]; } else { r = c = e - E; }
    float v = als[(size_t)r * H + hh] + ald[(size_t)c * H + hh];
    v = (v > 0.f) ? v : NEG_SLOPE * v;
    float exv = __expf(v - dec_f32(mmax[(size_t)c * H + hh]));
    ex[i] = exv;
    atomicAdd(&ssum[(size_t)c * H + hh], exv);
}

// one wave (32 lanes) per edge; float4 gathers, scalar f32 global atomic adds
__global__ void edge_agg(const float* __restrict__ hlin, const float* __restrict__ ex,
                         const float* __restrict__ ssum,
                         const int* __restrict__ row, const int* __restrict__ col,
                         float* __restrict__ agg, int E, int ET, int H, int D)
{
    long gtid = (long)blockIdx.x * blockDim.x + threadIdx.x;
    int wid  = (int)(gtid >> 5);
    int lane = (int)(gtid & 31);
    if (wid >= ET) return;
    int r, c;
    if (wid < E) { r = row[wid]; c = col[wid]; } else { r = c = wid - E; }
    const int F   = H * D;
    const int per = F >> 5;                 // 16 (H=4) or 4 (H=1)
    const int base = lane * per;
    const float* src = hlin + (size_t)r * F;
    float* dst = agg + (size_t)c * F;
#pragma unroll 4
    for (int i = 0; i < per; i += 4) {
        int e0 = base + i;
        int hh = e0 / D;
        float alpha = ex[(size_t)wid * H + hh] /
                      (ssum[(size_t)c * H + hh] + 1e-16f);
        float4 v = *(const float4*)(src + e0);
        atomicAdd(dst + e0 + 0, v.x * alpha);
        atomicAdd(dst + e0 + 1, v.y * alpha);
        atomicAdd(dst + e0 + 2, v.z * alpha);
        atomicAdd(dst + e0 + 3, v.w * alpha);
    }
}

// bias + optional ELU, writing f32 (layer 3 / output h)
__global__ void bias_act_f32(float* __restrict__ out, const float* __restrict__ in,
                             const float* __restrict__ bias, long n, int F, int elu)
{
    long i = (long)blockIdx.x * blockDim.x + threadIdx.x;
    if (i >= n) return;
    float v = in[i] + bias[(int)(i % F)];
    if (elu && v < 0.f) v = expm1f(v);
    out[i] = v;
}

// bias + ELU fused with bf16 convert (layers 1-2: activation feeds next GEMM only)
__global__ void bias_act_bf16(__bf16* __restrict__ out, const float* __restrict__ in,
                              const float* __restrict__ bias, long n, int F)
{
    long i = (long)blockIdx.x * blockDim.x + threadIdx.x;
    if (i >= n) return;
    float v = in[i] + bias[(int)(i % F)];
    if (v < 0.f) v = expm1f(v);
    out[i] = (__bf16)v;
}

__global__ void bn_cvt(const float* __restrict__ z,
                       const float* __restrict__ g, const float* __restrict__ b,
                       const float* __restrict__ rm, const float* __restrict__ rv,
                       __bf16* __restrict__ out, long n)
{
    long i = (long)blockIdx.x * blockDim.x + threadIdx.x;
    if (i >= n) return;
    int j = (int)(i & (HID - 1));
    float v = (z[i] - rm[j]) * rsqrtf(rv[j] + 1e-5f) * g[j] + b[j];
    out[i] = (__bf16)v;
}

// final 64->1 projections (verification: sigmoid, sequestration: linear)
__global__ void head1(const float* __restrict__ zin, const float* __restrict__ w,
                      const float* __restrict__ b, float* __restrict__ out,
                      int Nn, int do_sigmoid)
{
    int i = blockIdx.x * blockDim.x + threadIdx.x;
    if (i >= Nn) return;
    const float* z = zin + (size_t)i * 64;
    float a = b[0];
#pragma unroll 8
    for (int j = 0; j < 64; ++j) a += z[j] * w[j];
    out[i] = do_sigmoid ? (1.f / (1.f + __expf(-a))) : a;
}

// fraud = softmax(relu(Pr[row]+Pc[col]) @ W2 + b2)[:,1]
__global__ void fraud_kernel(const float* __restrict__ Pr, const float* __restrict__ Pc,
                             const int* __restrict__ row, const int* __restrict__ col,
                             const float* __restrict__ W2, const float* __restrict__ b2,
                             float* __restrict__ out, int E)
{
    int e = blockIdx.x * blockDim.x + threadIdx.x;
    if (e >= E) return;
    const float* pr = Pr + (size_t)row[e] * 64;
    const float* pc = Pc + (size_t)col[e] * 64;
    float l0 = b2[0], l1 = b2[1];
#pragma unroll 8
    for (int j = 0; j < 64; ++j) {
        float z = pr[j] + pc[j];
        z = (z > 0.f) ? z : 0.f;
        l0 += z * W2[2 * j];
        l1 += z * W2[2 * j + 1];
    }
    out[e] = 1.f / (1.f + __expf(l0 - l1));   // softmax[:,1]
}

// ---------------------------------------------------------------------------
// Host-side orchestration
// ---------------------------------------------------------------------------
static inline int cdiv(long a, long b) { return (int)((a + b - 1) / b); }

// One GAT layer. Abf must already hold the bf16 input activations [NN, Fin].
// If nextAbf != null the activation (bias+ELU) is written as bf16 there;
// otherwise written as f32 (no ELU) to outF32.
static void run_gat(hipStream_t s, int Fin,
                    const float* W, const float* a_s, const float* a_d,
                    const float* bias, int H, int D,
                    const int* row, const int* col,
                    __bf16* Abf, __bf16* Wbf, float* hlin, float* agg,
                    float* als, float* ald, unsigned* mmax, float* ssum, float* ex,
                    __bf16* nextAbf, float* outF32)
{
    const int F = H * D;
    long n;

    n = (long)Fin * F;  cvt_t_bf16<<<cdiv(n, 256), 256, 0, s>>>(W, Wbf, Fin, F);

    dim3 g(F / 64, (NN + 63) / 64);
    gemm_bf16_wmma<<<g, 128, 0, s>>>(Abf, Wbf, hlin, NN, F, Fin, nullptr, 0);

    n = (long)NN * H;
    attn_logits<<<cdiv(n, 256), 256, 0, s>>>(hlin, a_s, a_d, als, ald, NN, H, D);
    fill_u32<<<cdiv(n, 256), 256, 0, s>>>(mmax, ENC_NEG_INF, n);
    fill_u32<<<cdiv(n, 256), 256, 0, s>>>((uint32_t*)ssum, 0u, n);
    long na = (long)NN * F;
    fill_u32<<<cdiv(na, 256), 256, 0, s>>>((uint32_t*)agg, 0u, na);

    long ne = (long)ETOT * H;
    edge_max<<<cdiv(ne, 256), 256, 0, s>>>(row, col, als, ald, mmax, NE, ETOT, H);
    edge_exp<<<cdiv(ne, 256), 256, 0, s>>>(row, col, als, ald, mmax, ex, ssum, NE, ETOT, H);
    long nt = (long)ETOT * 32;
    edge_agg<<<cdiv(nt, 256), 256, 0, s>>>(hlin, ex, ssum, row, col, agg, NE, ETOT, H, D);

    if (nextAbf)
        bias_act_bf16<<<cdiv(na, 256), 256, 0, s>>>(nextAbf, agg, bias, na, F);
    else
        bias_act_f32<<<cdiv(na, 256), 256, 0, s>>>(outF32, agg, bias, na, F, 0);
}

extern "C" void kernel_launch(void* const* d_in, const int* in_sizes, int n_in,
                              void* d_out, int out_size, void* d_ws, size_t ws_size,
                              hipStream_t stream)
{
    // ---- inputs (setup_inputs dict order; params in insertion order) ----
    const float* x    = (const float*)d_in[0];
    const int*   ei   = (const int*)d_in[1];
    const int*   row  = ei;
    const int*   col  = ei + NE;
    const float* W1   = (const float*)d_in[2];
    const float* as1  = (const float*)d_in[3];
    const float* ad1  = (const float*)d_in[4];
    const float* b1   = (const float*)d_in[5];
    const float* W2   = (const float*)d_in[6];
    const float* as2  = (const float*)d_in[7];
    const float* ad2  = (const float*)d_in[8];
    const float* b2   = (const float*)d_in[9];
    const float* W3   = (const float*)d_in[10];
    const float* as3  = (const float*)d_in[11];
    const float* ad3  = (const float*)d_in[12];
    const float* b3   = (const float*)d_in[13];
    const float* v_w1 = (const float*)d_in[14];
    const float* v_b1 = (const float*)d_in[15];
    const float* bn_g = (const float*)d_in[16];
    const float* bn_b = (const float*)d_in[17];
    const float* bn_rm= (const float*)d_in[18];
    const float* bn_rv= (const float*)d_in[19];
    const float* v_w2 = (const float*)d_in[20];
    const float* v_b2 = (const float*)d_in[21];
    const float* v_w3 = (const float*)d_in[22];
    const float* v_b3 = (const float*)d_in[23];
    const float* s_w1 = (const float*)d_in[24];
    const float* s_b1 = (const float*)d_in[25];
    const float* s_w2 = (const float*)d_in[26];
    const float* s_b2 = (const float*)d_in[27];
    const float* f_w1 = (const float*)d_in[28];
    const float* f_b1 = (const float*)d_in[29];
    const float* f_w2 = (const float*)d_in[30];
    const float* f_b2 = (const float*)d_in[31];

    // ---- outputs: [ver N | seq N | fraud E | h N*128] ----
    float* o_ver   = (float*)d_out;
    float* o_seq   = o_ver + NN;
    float* o_fraud = o_seq + NN;
    float* o_h     = o_fraud + NE;     // h3 lives here and feeds the heads

    // ---- workspace layout ----
    float*    hlin = (float*)d_ws;                 // N x 512
    float*    agg  = hlin + (long)NN * 512;        // N x 512
    float*    als  = agg  + (long)NN * 512;        // N x 4
    float*    ald  = als  + (long)NN * HEADS;
    unsigned* mmax = (unsigned*)(ald + (long)NN * HEADS);
    float*    ssum = (float*)(mmax + (long)NN * HEADS);
    float*    ex   = ssum + (long)NN * HEADS;      // ETOT x 4
    __bf16*   Abf  = (__bf16*)(ex + (long)ETOT * HEADS);   // N x 512 bf16
    __bf16*   Wbf  = Abf + (long)NN * 512;                 // 512 x 512 bf16
    // head intermediates reuse the big f32 buffers:
    float* z1 = agg;                   // N x 128
    float* z2 = hlin;                  // N x 64
    float* s1 = hlin + (long)NN * 64;  // N x 64
    float* Pr = hlin + (long)NN * 128; // N x 64
    float* Pc = hlin + (long)NN * 192; // N x 64

    // ---- 3 GAT layers (activation stays bf16 between layers) ----
    long n = (long)NN * 96;
    cvt_bf16<<<cdiv(n, 256), 256, 0, stream>>>(x, Abf, n);
    run_gat(stream, 96,        W1, as1, ad1, b1, HEADS, HID, row, col,
            Abf, Wbf, hlin, agg, als, ald, mmax, ssum, ex, Abf, nullptr);
    run_gat(stream, HEADS*HID, W2, as2, ad2, b2, HEADS, HID, row, col,
            Abf, Wbf, hlin, agg, als, ald, mmax, ssum, ex, Abf, nullptr);
    run_gat(stream, HEADS*HID, W3, as3, ad3, b3, 1,     HID, row, col,
            Abf, Wbf, hlin, agg, als, ald, mmax, ssum, ex, nullptr, o_h);

    // ---- heads on h3 (in d_out) ----
    n = (long)NN * HID; cvt_bf16<<<cdiv(n, 256), 256, 0, stream>>>(o_h, Abf, n);

    // verification: z1 = relu(h @ v_w1 + v_b1)
    cvt_t_bf16<<<cdiv(128 * 128, 256), 256, 0, stream>>>(v_w1, Wbf, 128, 128);
    { dim3 g(2, (NN + 63) / 64);
      gemm_bf16_wmma<<<g, 128, 0, stream>>>(Abf, Wbf, z1, NN, 128, 128, v_b1, 1); }

    // sequestration: s1 = relu(h @ s_w1 + s_b1)
    cvt_t_bf16<<<cdiv(128 * 64, 256), 256, 0, stream>>>(s_w1, Wbf, 128, 64);
    { dim3 g(1, (NN + 63) / 64);
      gemm_bf16_wmma<<<g, 128, 0, stream>>>(Abf, Wbf, s1, NN, 64, 128, s_b1, 1); }

    // fraud projections: Pr = h @ f_w1[:128] + f_b1, Pc = h @ f_w1[128:]
    cvt_t_bf16<<<cdiv(128 * 64, 256), 256, 0, stream>>>(f_w1, Wbf, 128, 64);
    cvt_t_bf16<<<cdiv(128 * 64, 256), 256, 0, stream>>>(f_w1 + 128 * 64,
                                                        Wbf + 64 * 128, 128, 64);
    { dim3 g(1, (NN + 63) / 64);
      gemm_bf16_wmma<<<g, 128, 0, stream>>>(Abf, Wbf, Pr, NN, 64, 128, f_b1, 0);
      gemm_bf16_wmma<<<g, 128, 0, stream>>>(Abf, Wbf + 64 * 128, Pc, NN, 64, 128,
                                            nullptr, 0); }

    // batchnorm(z1) -> bf16 (overwrites Abf; all Abf consumers are done)
    n = (long)NN * HID;
    bn_cvt<<<cdiv(n, 256), 256, 0, stream>>>(z1, bn_g, bn_b, bn_rm, bn_rv, Abf, n);

    // z2 = relu(z1bn @ v_w2 + v_b2)
    cvt_t_bf16<<<cdiv(128 * 64, 256), 256, 0, stream>>>(v_w2, Wbf, 128, 64);
    { dim3 g(1, (NN + 63) / 64);
      gemm_bf16_wmma<<<g, 128, 0, stream>>>(Abf, Wbf, z2, NN, 64, 128, v_b2, 1); }

    // final tiny projections + edge fraud head
    head1<<<cdiv(NN, 256), 256, 0, stream>>>(z2, v_w3, v_b3, o_ver, NN, 1);
    head1<<<cdiv(NN, 256), 256, 0, stream>>>(s1, s_w2, s_b2, o_seq, NN, 0);
    fraud_kernel<<<cdiv(NE, 256), 256, 0, stream>>>(Pr, Pc, row, col, f_w2, f_b2,
                                                    o_fraud, NE);
}
